// FusedMoE_71399536328817
// MI455X (gfx1250) — matrix operand order repaired and checked by reading, not compiled, not run on hardware
//
#include <hip/hip_runtime.h>

// Problem constants (match reference): T=1024, E=8, H=1024, I=2816, TOP_K=2
#define T_TOK 1024
#define NE    8
#define HD    1024
#define ID    2816
#define NSLOT (2 * T_TOK)   // every token lands in exactly 2 expert slots

typedef __attribute__((ext_vector_type(16))) __bf16 bf16x16;
typedef __attribute__((ext_vector_type(8)))  __bf16 bf16x8;
typedef __attribute__((ext_vector_type(8)))  float  f32x8;

// GEMM tiling: 64x64 block tile, BK=64 (two 16x16x32 WMMA k-steps per stage),
// 8 wave32 waves arranged 4(M) x 2(N); each wave owns a 16x32 C tile.
// LDS tiles are double buffered; one barrier per K-tile.
#define BM  64
#define BN  64
#define BK  64
#define LDA 80   // padded LDS row stride in bf16 elems (160B rows, 32B aligned)

__device__ __forceinline__ void wait_async0()
{
#if __has_builtin(__builtin_amdgcn_s_wait_asynccnt)
    __builtin_amdgcn_s_wait_asynccnt(0);
#else
    asm volatile("s_wait_asynccnt 0x0" ::: "memory");
#endif
}

// ---------------------------------------------------------------------------
// Kernel 1: router. softmax + top-2 per token, grouped scatter by expert.
// ---------------------------------------------------------------------------
__global__ __launch_bounds__(T_TOK)
void moe_router(const float* __restrict__ logits,
                int*   __restrict__ expert_off,   // E+1 exclusive offsets
                int*   __restrict__ slot_token,   // slot -> token id
                int*   __restrict__ token_slots,  // token -> its 2 slots
                float* __restrict__ slot_w)       // slot -> routing weight
{
    __shared__ int cnt[NE];
    __shared__ int offs[NE + 1];
    const int t = threadIdx.x;
    if (t < NE) cnt[t] = 0;
    __syncthreads();

    float l[NE];
#pragma unroll
    for (int e = 0; e < NE; ++e) l[e] = logits[t * NE + e];

    // top-1 (lowest index wins ties, like jax.lax.top_k)
    int e1 = 0;
#pragma unroll
    for (int e = 1; e < NE; ++e) if (l[e] > l[e1]) e1 = e;
    // top-2
    int e2 = (e1 == 0) ? 1 : 0;
#pragma unroll
    for (int e = 0; e < NE; ++e) if (e != e1 && l[e] > l[e2]) e2 = e;

    // normalized top-2 softmax weights: common denominator cancels
    const float p2  = __expf(l[e2] - l[e1]);
    const float inv = __builtin_amdgcn_rcpf(1.0f + p2);
    const float w1  = inv;
    const float w2v = p2 * inv;

    const int pos1 = atomicAdd(&cnt[e1], 1);
    const int pos2 = atomicAdd(&cnt[e2], 1);
    __syncthreads();

    if (t == 0) {
        int acc = 0;
        for (int e = 0; e < NE; ++e) { offs[e] = acc; acc += cnt[e]; }
        offs[NE] = acc;   // == NSLOT
        for (int e = 0; e <= NE; ++e) expert_off[e] = offs[e];
    }
    __syncthreads();

    const int s1 = offs[e1] + pos1;
    const int s2 = offs[e2] + pos2;
    slot_token[s1] = t;  slot_w[s1] = w1;   token_slots[2 * t + 0] = s1;
    slot_token[s2] = t;  slot_w[s2] = w2v;  token_slots[2 * t + 1] = s2;
}

// ---------------------------------------------------------------------------
// LDS fragment helpers matching CDNA5 WMMA bf16 per-lane layouts (ISA 7.12.2).
// A (16x32 M×K): lane (hi,lmod): elems 0-7 -> K = kk + hi*8 + j,
//                                elems 8-15 -> K = kk + 16 + hi*8 + (j-8)
// B (32x16 K×N): lane (hi,lmod): same K order as A, N = lmod
// ---------------------------------------------------------------------------
__device__ __forceinline__ bf16x16 load_frag_a(const __bf16* s, int m, int kk, int lhalf)
{
    const bf16x8 lo = *(const bf16x8*)&s[m * LDA + kk + lhalf * 8];
    const bf16x8 hi = *(const bf16x8*)&s[m * LDA + kk + 16 + lhalf * 8];
    bf16x16 a;
#pragma unroll
    for (int j = 0; j < 8; ++j) { a[j] = lo[j]; a[8 + j] = hi[j]; }
    return a;
}
__device__ __forceinline__ bf16x16 load_frag_b(const __bf16* s, int n, int kk, int lhalf)
{
    const bf16x8 lo = *(const bf16x8*)&s[n * LDA + kk + lhalf * 8];
    const bf16x8 hi = *(const bf16x8*)&s[n * LDA + kk + 16 + lhalf * 8];
    bf16x16 b;
#pragma unroll
    for (int j = 0; j < 8; ++j) { b[j] = lo[j]; b[8 + j] = hi[j]; }
    return b;
}
__device__ __forceinline__ void cvt_store16(__bf16* dst, const float4* r)
{
#pragma unroll
    for (int j = 0; j < 4; ++j) {
        dst[4*j+0] = (__bf16)r[j].x; dst[4*j+1] = (__bf16)r[j].y;
        dst[4*j+2] = (__bf16)r[j].z; dst[4*j+3] = (__bf16)r[j].w;
    }
}

// ---------------------------------------------------------------------------
// Kernel 2: grouped gate_up GEMM + SiLU, double-buffered LDS pipeline.
//   act[slot, i] = route_w[slot] * silu(x·w13[e,i,:]) * (x·w13[e,I+i,:])
// ---------------------------------------------------------------------------
__global__ __launch_bounds__(256)
void moe_gateup(const float* __restrict__ hidden,
                const float* __restrict__ w13,
                const int*   __restrict__ expert_off,
                const int*   __restrict__ slot_token,
                const float* __restrict__ slot_w,
                __bf16*      __restrict__ act)
{
    const int e   = blockIdx.z;
    const int off = expert_off[e];
    const int cnt = expert_off[e + 1] - off;
    const int r0  = blockIdx.y * BM;
    if (r0 >= cnt) return;                       // uniform early-exit
    const int n0  = blockIdx.x * BN;

    __shared__ __attribute__((aligned(32))) __bf16 sX[2][BM * LDA];
    __shared__ __attribute__((aligned(32))) __bf16 sG[2][BN * LDA];
    __shared__ __attribute__((aligned(32))) __bf16 sU[2][BN * LDA];
    __shared__ int   sTok[BM];
    __shared__ float sW[BM];

    const int tid = threadIdx.x;
    if (tid < BM) {
        const int r = r0 + tid;
        sTok[tid] = (r < cnt) ? slot_token[off + r] : 0;
        sW[tid]   = (r < cnt) ? slot_w[off + r]     : 0.0f;
    }
    __syncthreads();

    const float* __restrict__ w13e = w13 + (size_t)e * (2 * ID) * HD;

    const int wave  = tid >> 5;
    const int lane  = tid & 31;
    const int wm    = wave >> 1;     // 0..3 -> M subtile
    const int wn    = wave & 1;      // 0..1 -> N half
    const int lmod  = lane & 15;
    const int lhalf = lane >> 4;

    // staging: 256 threads, each moves 16 fp32 per tile per stage
    const int row = tid >> 2;          // 0..63
    const int c0  = (tid & 3) * 16;    // 0,16,32,48
    const int tok = sTok[row];
    const float* xsrc = hidden + (size_t)tok * HD + c0;
    const float* gsrc = w13e + (size_t)(n0 + row) * HD + c0;
    const float* usrc = w13e + (size_t)(ID + n0 + row) * HD + c0;

    float4 rx[4], rg[4], ru[4];
    auto stage_load = [&](int k0) {
#pragma unroll
        for (int j = 0; j < 4; ++j) {
            rx[j] = *(const float4*)(xsrc + k0 + 4 * j);
            rg[j] = *(const float4*)(gsrc + k0 + 4 * j);
            ru[j] = *(const float4*)(usrc + k0 + 4 * j);
        }
    };
    auto stage_store = [&](int b) {
        cvt_store16(&sX[b][row * LDA + c0], rx);
        cvt_store16(&sG[b][row * LDA + c0], rg);
        cvt_store16(&sU[b][row * LDA + c0], ru);
    };

    f32x8 accG[2] = {f32x8{0.f}, f32x8{0.f}};
    f32x8 accU[2] = {f32x8{0.f}, f32x8{0.f}};

    stage_load(0);
    stage_store(0);
    __syncthreads();

    int buf = 0;
    for (int k0 = 0; k0 < HD; k0 += BK) {
        const int nk = k0 + BK;
        if (nk < HD) stage_load(nk);             // prefetch next tile -> regs

        const __bf16* bx = sX[buf];
        const __bf16* bg = sG[buf];
        const __bf16* bu = sU[buf];
#pragma unroll
        for (int kk = 0; kk < BK; kk += 32) {
            const bf16x16 a = load_frag_a(bx, wm * 16 + lmod, kk, lhalf);
#pragma unroll
            for (int s = 0; s < 2; ++s) {
                const int n = wn * 32 + s * 16 + lmod;
                const bf16x16 fg = load_frag_b(bg, n, kk, lhalf);
                const bf16x16 fu = load_frag_b(bu, n, kk, lhalf);
                accG[s] = __builtin_amdgcn_wmma_f32_16x16x32_bf16(
                    false, a, false, fg, (short)0, accG[s], false, false);
                accU[s] = __builtin_amdgcn_wmma_f32_16x16x32_bf16(
                    false, a, false, fu, (short)0, accU[s], false, false);
            }
        }
        if (nk < HD) {
            stage_store(buf ^ 1);                // fill other buffer
            __syncthreads();                     // one barrier per K-tile
        }
        buf ^= 1;
    }

    // epilogue: act = route_w * silu(g) * u, stored bf16 (feeds down-GEMM A).
    // silu via v_rcp_f32 instead of the IEEE divide ladder.
#pragma unroll
    for (int s = 0; s < 2; ++s) {
        const int col = n0 + wn * 32 + s * 16 + lmod;
#pragma unroll
        for (int r = 0; r < 8; ++r) {
            const int mrow = wm * 16 + 8 * lhalf + r;   // C layout: M=r(+8 hi half)
            const int grow = r0 + mrow;
            if (grow < cnt) {
                const float g   = accG[s][r];
                const float u   = accU[s][r];
                const float sig = __builtin_amdgcn_rcpf(1.0f + __expf(-g));
                const float a   = g * sig * u * sW[mrow];
                act[(size_t)(off + grow) * ID + col] = (__bf16)a;
            }
        }
    }
}

// ---------------------------------------------------------------------------
// Kernel 3: grouped down GEMM, double-buffered. A (bf16 act) is copied with
// CDNA5 async global->LDS (ASYNCcnt); B (fp32 w2) is reg-staged + converted.
//   down_buf[slot, h] = act[slot,:] · w2[e,h,:]
// ---------------------------------------------------------------------------
__global__ __launch_bounds__(256)
void moe_down(const __bf16* __restrict__ act,
              const float*  __restrict__ w2,
              const int*    __restrict__ expert_off,
              float*        __restrict__ down_buf)
{
    const int e   = blockIdx.z;
    const int off = expert_off[e];
    const int cnt = expert_off[e + 1] - off;
    const int r0  = blockIdx.y * BM;
    if (r0 >= cnt) return;
    const int h0  = blockIdx.x * BN;

    __shared__ __attribute__((aligned(32))) __bf16 sA[2][BM * LDA];
    __shared__ __attribute__((aligned(32))) __bf16 sB[2][BN * LDA];

    const int tid   = threadIdx.x;
    const int wave  = tid >> 5;
    const int lane  = tid & 31;
    const int wm    = wave >> 1;
    const int wn    = wave & 1;
    const int lmod  = lane & 15;
    const int lhalf = lane >> 4;

    const int row = tid >> 2;
    const int c0  = (tid & 3) * 16;
    int rr = r0 + row; if (rr > cnt - 1) rr = cnt - 1;      // clamp OOB rows
    const __bf16* asrc = act + (size_t)(off + rr) * ID + c0;
    const float*  bsrc = w2 + (size_t)e * HD * ID + (size_t)(h0 + row) * ID + c0;

    // A tile: async copy straight into LDS (32B per thread, offset hits both
    // the LDS and global address per the async addressing rules).
    auto a_async = [&](int k0, int b) {
        const unsigned laddr = (unsigned)(uintptr_t)&sA[b][row * LDA + c0];
        const __bf16* g = asrc + k0;
        asm volatile("global_load_async_to_lds_b128 %0, %1, off\n\t"
                     "global_load_async_to_lds_b128 %0, %1, off offset:16"
                     :: "v"(laddr), "v"(g) : "memory");
    };

    float4 rb[4];
    auto b_load = [&](int k0) {
#pragma unroll
        for (int j = 0; j < 4; ++j) rb[j] = *(const float4*)(bsrc + k0 + 4 * j);
    };
    auto b_store = [&](int b) { cvt_store16(&sB[b][row * LDA + c0], rb); };

    f32x8 acc[2] = {f32x8{0.f}, f32x8{0.f}};

    a_async(0, 0);
    b_load(0);
    b_store(0);
    wait_async0();
    __syncthreads();

    int buf = 0;
    for (int k0 = 0; k0 < ID; k0 += BK) {
        const int nk = k0 + BK;
        if (nk < ID) {
            a_async(nk, buf ^ 1);               // async fill of other buffer
            b_load(nk);                          // prefetch B -> regs
        }

        const __bf16* ba = sA[buf];
        const __bf16* bb = sB[buf];
#pragma unroll
        for (int kk = 0; kk < BK; kk += 32) {
            const bf16x16 a = load_frag_a(ba, wm * 16 + lmod, kk, lhalf);
#pragma unroll
            for (int s = 0; s < 2; ++s) {
                const bf16x16 b = load_frag_b(bb, wn * 32 + s * 16 + lmod, kk, lhalf);
                acc[s] = __builtin_amdgcn_wmma_f32_16x16x32_bf16(
                    false, a, false, b, (short)0, acc[s], false, false);
            }
        }
        if (nk < ID) {
            b_store(buf ^ 1);
            wait_async0();                       // A tile landed in LDS
            __syncthreads();
        }
        buf ^= 1;
    }

#pragma unroll
    for (int s = 0; s < 2; ++s) {
        const int col = h0 + wn * 32 + s * 16 + lmod;
#pragma unroll
        for (int r = 0; r < 8; ++r) {
            const int mrow = wm * 16 + 8 * lhalf + r;
            const int grow = r0 + mrow;
            if (grow < cnt)
                down_buf[(size_t)(off + grow) * HD + col] = acc[s][r];
        }
    }
}

// ---------------------------------------------------------------------------
// Kernel 4: combine. out[t,:] = down_buf[slot1(t),:] + down_buf[slot2(t),:]
// Deterministic fp32 sum (fixed order), float4-vectorized.
// ---------------------------------------------------------------------------
__global__ __launch_bounds__(256)
void moe_combine(const float* __restrict__ down_buf,
                 const int*   __restrict__ token_slots,
                 float*       __restrict__ out)
{
    const int i  = blockIdx.x * blockDim.x + threadIdx.x;  // over T*H/4
    const int t  = i >> 8;            // H/4 == 256
    const int c4 = (i & 255) * 4;
    const int s1 = token_slots[2 * t + 0];
    const int s2 = token_slots[2 * t + 1];
    const float4 a = *(const float4*)(down_buf + (size_t)s1 * HD + c4);
    const float4 b = *(const float4*)(down_buf + (size_t)s2 * HD + c4);
    float4 o;
    o.x = a.x + b.x; o.y = a.y + b.y; o.z = a.z + b.z; o.w = a.w + b.w;
    *(float4*)(out + (size_t)t * HD + c4) = o;
}

// ---------------------------------------------------------------------------
extern "C" void kernel_launch(void* const* d_in, const int* in_sizes, int n_in,
                              void* d_out, int out_size, void* d_ws, size_t ws_size,
                              hipStream_t stream)
{
    const float* hidden = (const float*)d_in[0];   // (T, H)
    const float* logits = (const float*)d_in[1];   // (T, E)
    const float* w13    = (const float*)d_in[2];   // (E, 2I, H)
    const float* w2     = (const float*)d_in[3];   // (E, H, I)
    float* out = (float*)d_out;                    // (T, H)

    // workspace carve-up (~20 MB total)
    char* ws = (char*)d_ws;
    int*    expert_off  = (int*)ws;      ws += 16 * sizeof(int);
    int*    slot_token  = (int*)ws;      ws += NSLOT * sizeof(int);
    int*    token_slots = (int*)ws;      ws += NSLOT * sizeof(int);
    float*  slot_w      = (float*)ws;    ws += NSLOT * sizeof(float);
    __bf16* act         = (__bf16*)ws;   ws += (size_t)NSLOT * ID * sizeof(__bf16);
    float*  down_buf    = (float*)ws;    // NSLOT * HD floats

    moe_router<<<1, T_TOK, 0, stream>>>(logits, expert_off, slot_token,
                                        token_slots, slot_w);
    moe_gateup<<<dim3(ID / BN, T_TOK / BM, NE), 256, 0, stream>>>(
        hidden, w13, expert_off, slot_token, slot_w, act);
    moe_down<<<dim3(HD / BN, T_TOK / BM, NE), 256, 0, stream>>>(
        act, w2, expert_off, down_buf);
    moe_combine<<<(T_TOK * HD / 4) / 256, 256, 0, stream>>>(
        down_buf, token_slots, out);
}
